// MRConv4d_68719476753
// MI455X (gfx1250) — compile-verified, matrix operand order checked by reading, not run on hardware
//
#include <hip/hip_runtime.h>
#include <hip/hip_bf16.h>
#include <math.h>

typedef __attribute__((ext_vector_type(16))) __bf16 v16bf;
typedef __attribute__((ext_vector_type(8)))  __bf16 v8bf;
typedef __attribute__((ext_vector_type(8)))  float  v8f;

#define CI   192
#define CO   384
#define HW   3136          // 56*56
#define BB   32
#define MTOT 100352        // BB*HW
#define KTOT 384

// ---------------- Kernel 1: pack W to bf16, zero BN accumulators ----------------
__global__ __launch_bounds__(256) void prep_kernel(const float* __restrict__ w,
                                                   __bf16* __restrict__ Bw,
                                                   float* __restrict__ sums,
                                                   float* __restrict__ sumsq) {
    int i = blockIdx.x * 256 + threadIdx.x;
    if (i < CO * KTOT) Bw[i] = (__bf16)w[i];
    if (i < CO) { sums[i] = 0.f; sumsq[i] = 0.f; }
}

// ---------------- Kernel 2: x_j via exclusive parity mins, pack A (bf16) --------
__global__ __launch_bounds__(256) void xj_pack_kernel(const float* __restrict__ x,
                                                      __bf16* __restrict__ A) {
    int bc = blockIdx.x;               // bc = b*192 + c
    int b  = bc / CI;
    int c  = bc % CI;
    __shared__ float sp[HW];
    __shared__ float cmin1[2][56], cmin2[2][56];   // along H per (parity, w)
    __shared__ float rmin1[2][56], rmin2[2][56];   // along W per (parity, h)

    const float* plane = x + (size_t)bc * HW;
    for (int i = threadIdx.x; i < HW; i += 256) sp[i] = plane[i];
    __syncthreads();

    if (threadIdx.x < 112) {
        int w = threadIdx.x >> 1, par = threadIdx.x & 1;
        float m1 = 3.0e38f, m2 = 3.0e38f;
        for (int h = par; h < 56; h += 2) {
            float v = sp[h * 56 + w];
            if (v < m1) { m2 = m1; m1 = v; } else if (v < m2) m2 = v;
        }
        cmin1[par][w] = m1; cmin2[par][w] = m2;
    } else if (threadIdx.x < 224) {
        int t = threadIdx.x - 112;
        int h = t >> 1, par = t & 1;
        float m1 = 3.0e38f, m2 = 3.0e38f;
        for (int w = par; w < 56; w += 2) {
            float v = sp[h * 56 + w];
            if (v < m1) { m2 = m1; m1 = v; } else if (v < m2) m2 = v;
        }
        rmin1[par][h] = m1; rmin2[par][h] = m2;
    }
    __syncthreads();

    size_t mbase = (size_t)b * HW;
    for (int e = threadIdx.x; e < HW; e += 256) {
        int h = e / 56, w = e - h * 56;
        float v = sp[e];
        float ch = cmin1[h & 1][w];  if (v == ch) ch = cmin2[h & 1][w];  // exclude self
        float rw = rmin1[w & 1][h];  if (v == rw) rw = rmin2[w & 1][h];
        float xj = v - fminf(ch, rw);
        xj = fmaxf(xj, 0.f);
        size_t row = (mbase + e) * (size_t)KTOT;
        A[row + c]        = (__bf16)v;
        A[row + CI + c]   = (__bf16)xj;
    }
}

// ---------------- Kernel 3: WMMA GEMM + fused BN-stat reduction ------------------
// Block tile 128x128, BK=32, 8 wave32 waves (4 in M x 2 in N), wave tile 32x64.
// Double-buffered LDS fed by global_load_async_to_lds_b128 (ASYNCcnt).
__global__ __launch_bounds__(256) void gemm_kernel(const __bf16* __restrict__ A,
                                                   const __bf16* __restrict__ Bw,
                                                   float* __restrict__ Y,
                                                   float* __restrict__ sums,
                                                   float* __restrict__ sumsq) {
    __shared__ __bf16 As[2][128][40];  // stride 40 bf16 (80B) -> conflict-free b128 reads
    __shared__ __bf16 Bs[2][128][40];  // stored n-major: Bs[buf][n][k]
    __shared__ float  red[128][2];     // per-block channel sum / sumsq

    const int m0  = blockIdx.x * 128;
    const int n0  = blockIdx.y * 128;
    const int tid = threadIdx.x;
    const int wv  = tid >> 5, ln = tid & 31;
    const int wvM = wv & 3,   wvN = wv >> 2;
    const int lr  = ln & 15,  hi  = ln >> 4;
    const int kk  = hi * 8;   // bf16 A/B lane layout: lanes 0-15 K=0..7/16..23, 16-31 K=8..15/24..31

    if (tid < 128) { red[tid][0] = 0.f; red[tid][1] = 0.f; }

    // Async tile fetch: each thread moves one 16B chunk of A and B per half,
    // 4 async ops per wave per tile -> ASYNCcnt += 4.
    auto issue_tile = [&](int kt, int buf) {
#pragma unroll
        for (int i = 0; i < 2; i++) {
            int idx = tid + i * 256;
            int r = idx >> 2, c8 = (idx & 3) * 8;
            unsigned long long ga = (unsigned long long)(size_t)&A [(size_t)(m0 + r) * KTOT + kt * 32 + c8];
            unsigned long long gb = (unsigned long long)(size_t)&Bw[(size_t)(n0 + r) * KTOT + kt * 32 + c8];
            unsigned int la = (unsigned int)(size_t)&As[buf][r][c8];  // flat addr[31:0] == LDS byte offset
            unsigned int lb = (unsigned int)(size_t)&Bs[buf][r][c8];
            asm volatile("global_load_async_to_lds_b128 %0, %1, off" :: "v"(la), "v"(ga) : "memory");
            asm volatile("global_load_async_to_lds_b128 %0, %1, off" :: "v"(lb), "v"(gb) : "memory");
        }
    };

    v8f acc[2][4];
#pragma unroll
    for (int i = 0; i < 2; i++)
#pragma unroll
        for (int j = 0; j < 4; j++) acc[i][j] = (v8f){0,0,0,0,0,0,0,0};

    issue_tile(0, 0);

    for (int kt = 0; kt < KTOT / 32; ++kt) {
        const int cur = kt & 1;
        if (kt + 1 < KTOT / 32) {
            issue_tile(kt + 1, cur ^ 1);                       // prefetch next tile
            asm volatile("s_wait_asynccnt 0x4" ::: "memory");  // oldest 4 (current tile) done
        } else {
            asm volatile("s_wait_asynccnt 0x0" ::: "memory");
        }
        __syncthreads();   // block-wide: current tile resident in LDS

        v16bf af[2], bfm[4];
#pragma unroll
        for (int tm = 0; tm < 2; tm++) {
            int row = wvM * 32 + tm * 16 + lr;
            v8bf lo  = *(const v8bf*)&As[cur][row][kk];
            v8bf hi8 = *(const v8bf*)&As[cur][row][kk + 16];
            af[tm] = __builtin_shufflevector(lo, hi8, 0,1,2,3,4,5,6,7,8,9,10,11,12,13,14,15);
        }
#pragma unroll
        for (int tn = 0; tn < 4; tn++) {
            int row = wvN * 64 + tn * 16 + lr;
            v8bf lo  = *(const v8bf*)&Bs[cur][row][kk];
            v8bf hi8 = *(const v8bf*)&Bs[cur][row][kk + 16];
            bfm[tn] = __builtin_shufflevector(lo, hi8, 0,1,2,3,4,5,6,7,8,9,10,11,12,13,14,15);
        }
#pragma unroll
        for (int tm = 0; tm < 2; tm++)
#pragma unroll
            for (int tn = 0; tn < 4; tn++)
                acc[tm][tn] = __builtin_amdgcn_wmma_f32_16x16x32_bf16(
                    false, af[tm], false, bfm[tn], (short)0, acc[tm][tn], false, false);
        __syncthreads();   // done reading buf[cur] before it is refilled
    }

    // Epilogue: C layout -> lane holds n = lr, rows m = base + hi*8 + r (r = vgpr idx).
    // Y layout [b][o][hw]; 16-row tiles never cross b (3136 % 16 == 0).
#pragma unroll
    for (int tm = 0; tm < 2; tm++) {
        int mb   = m0 + wvM * 32 + tm * 16 + hi * 8;
        int bidx = mb / HW;
        int hw   = mb - bidx * HW;
#pragma unroll
        for (int tn = 0; tn < 4; tn++) {
            int cn = wvN * 64 + tn * 16 + lr;       // block-local channel
            float* dst = Y + ((size_t)bidx * CO + (n0 + cn)) * HW + hw;
            v8f a = acc[tm][tn];
            *(float4*)(dst)     = make_float4(a[0], a[1], a[2], a[3]);
            *(float4*)(dst + 4) = make_float4(a[4], a[5], a[6], a[7]);
            float s  = a[0] + a[1] + a[2] + a[3] + a[4] + a[5] + a[6] + a[7];
            float s2 = a[0]*a[0] + a[1]*a[1] + a[2]*a[2] + a[3]*a[3]
                     + a[4]*a[4] + a[5]*a[5] + a[6]*a[6] + a[7]*a[7];
            atomicAdd(&red[cn][0], s);              // ds_add_f32
            atomicAdd(&red[cn][1], s2);
        }
    }
    __syncthreads();
    if (tid < 128) {
        atomicAdd(&sums [n0 + tid], red[tid][0]);   // global_atomic_add_f32
        atomicAdd(&sumsq[n0 + tid], red[tid][1]);
    }
}

// ---------------- Kernel 4: per-channel scale/shift ------------------------------
__global__ void bnscale_kernel(const float* __restrict__ sums, const float* __restrict__ sumsq,
                               const float* __restrict__ gamma, const float* __restrict__ beta,
                               float* __restrict__ sc, float* __restrict__ sh) {
    int o = blockIdx.x * 64 + threadIdx.x;
    if (o < CO) {
        const float invN = 1.f / (float)MTOT;
        float mean = sums[o] * invN;
        float var  = sumsq[o] * invN - mean * mean;
        float s = gamma[o] * rsqrtf(var + 1e-5f);
        sc[o] = s;
        sh[o] = beta[o] - mean * s;
    }
}

// ---------------- Kernel 5: fused BN apply + exact-erf GELU (in place) ----------
__global__ __launch_bounds__(256) void bngelu_kernel(float* __restrict__ Y,
                                                     const float* __restrict__ sc,
                                                     const float* __restrict__ sh) {
    size_t idx = (size_t)blockIdx.x * 256 + threadIdx.x;
    int o = (int)((idx / HW) % CO);
    float v = Y[idx] * sc[o] + sh[o];
    Y[idx] = 0.5f * v * (1.f + erff(v * 0.70710678118654752f));
}

extern "C" void kernel_launch(void* const* d_in, const int* in_sizes, int n_in,
                              void* d_out, int out_size, void* d_ws, size_t ws_size,
                              hipStream_t stream) {
    (void)in_sizes; (void)n_in; (void)out_size; (void)ws_size;
    const float* x     = (const float*)d_in[0];
    const float* w     = (const float*)d_in[1];
    // d_in[2] (conv bias) is mathematically cancelled by training-mode BatchNorm
    // (a per-channel additive constant shifts the batch mean by exactly itself).
    const float* gamma = (const float*)d_in[3];
    const float* beta  = (const float*)d_in[4];

    char*   ws    = (char*)d_ws;
    __bf16* A     = (__bf16*)ws;                      // 100352*384*2 = 77,070,336 B
    __bf16* Bw    = (__bf16*)(ws + 77070336);         //    384*384*2 =    294,912 B
    float*  sums  = (float*)(ws + 77365248);
    float*  sumsq = sums + CO;
    float*  scale = sums + 2 * CO;
    float*  shift = sums + 3 * CO;
    float*  Y     = (float*)d_out;

    prep_kernel   <<<(CO * KTOT + 255) / 256, 256, 0, stream>>>(w, Bw, sums, sumsq);
    xj_pack_kernel<<<BB * CI, 256, 0, stream>>>(x, A);
    gemm_kernel   <<<dim3(MTOT / 128, CO / 128), 256, 0, stream>>>(A, Bw, Y, sums, sumsq);
    bnscale_kernel<<<6, 64, 0, stream>>>(sums, sumsq, gamma, beta, scale, shift);
    bngelu_kernel <<<(MTOT * CO) / 256, 256, 0, stream>>>(Y, scale, shift);
}
